// SAModule_63419487093386
// MI455X (gfx1250) — matrix operand order repaired and checked by reading, not compiled, not run on hardware
//
#include <hip/hip_runtime.h>
#include <hip/hip_bf16.h>

#define B_   8
#define N_   4096
#define D_   64
#define S_   1024
#define K_   64
#define H_   64
#define CO_  128
#define R2_  0.04f
#define GPB_ 4   // groups per block in the MLP kernel

typedef __attribute__((ext_vector_type(16))) __bf16 v16bf;
typedef __attribute__((ext_vector_type(8)))  float  v8f;

// ---------------- fragment loads from LDS (16-bit A/B layouts per CDNA5 ISA 7.12.2) ---
// A 16x32 bf16: lane m=l&15, half h=l>>4; lane holds K in [8h,8h+8) and [16+8h,16+8h+8)
__device__ __forceinline__ v16bf lds_frag_a(const __bf16* row, int k0, int h) {
    union { uint4 u[2]; v16bf v; } c;
    c.u[0] = *(const uint4*)(row + k0 + 8 * h);
    c.u[1] = *(const uint4*)(row + k0 + 16 + 8 * h);
    return c.v;
}
// B 32x16 bf16 (weights stored transposed [N][K]): lane n=l&15, holds K in [8h,8h+8) and [16+8h,16+8h+8)
__device__ __forceinline__ v16bf lds_frag_b(const __bf16* row, int k0, int h) {
    union { uint4 u[2]; v16bf v; } c;
    c.u[0] = *(const uint4*)(row + k0 + 8 * h);
    c.u[1] = *(const uint4*)(row + k0 + 16 + 8 * h);
    return c.v;
}

// CDNA5 async global->LDS copy (16B per lane), tracked on ASYNCcnt (ISA 10. / 15.18.3)
__device__ __forceinline__ void async_g2l_b128(unsigned lds_off, const void* gptr) {
    asm volatile("global_load_async_to_lds_b128 %0, %1, off"
                 :: "v"(lds_off), "v"((unsigned long long)(size_t)gptr)
                 : "memory");
}
__device__ __forceinline__ void wait_asynccnt0() {
    asm volatile("s_wait_asynccnt 0" ::: "memory");
}
__device__ __forceinline__ unsigned lds_addr_of(const void* p) {
    // flat address of a __shared__ object: low 32 bits are the LDS byte address
    return (unsigned)(size_t)p;
}

// ---------------- weight conversion fp32 -> bf16, transposed + K-padded -------------
// w1t[64][96] (n,k), w2t[64][64], w3t[128][64]
__global__ void convw_kernel(const float* __restrict__ W1, const float* __restrict__ W2,
                             const float* __restrict__ W3,
                             __bf16* __restrict__ w1t, __bf16* __restrict__ w2t,
                             __bf16* __restrict__ w3t) {
    int i = blockIdx.x * 256 + threadIdx.x;
    if (i < 6144) {
        int n = i / 96, k = i % 96;
        w1t[i] = (__bf16)((k < 67) ? W1[k * 64 + n] : 0.0f);
    } else if (i < 10240) {
        int j = i - 6144; int n = j / 64, k = j % 64;
        w2t[j] = (__bf16)W2[k * 64 + n];
    } else if (i < 18432) {
        int j = i - 10240; int n = j / 64, k = j % 64;
        w3t[j] = (__bf16)W3[k * 128 + n];
    }
}

// ---------------- farthest point sampling: 1 block (1024 thr) per batch -------------
__global__ __launch_bounds__(1024) void fps_kernel(const float* __restrict__ pos,
                                                   float* __restrict__ newxyz) {
    int b = blockIdx.x;
    int tid = threadIdx.x, lane = tid & 31, wid = tid >> 5;
    __shared__ float sv[32];
    __shared__ int   si[32];
    __shared__ float slast[3];

    float px[4], py[4], pz[4], mind[4];
#pragma unroll
    for (int j = 0; j < 4; ++j) {
        int p = tid + 1024 * j;
        const float* pp = pos + ((size_t)b * N_ + p) * 3;
        px[j] = pp[0]; py[j] = pp[1]; pz[j] = pp[2];
        mind[j] = 1e10f;
    }
    float lx = pos[(size_t)b * N_ * 3 + 0];
    float ly = pos[(size_t)b * N_ * 3 + 1];
    float lz = pos[(size_t)b * N_ * 3 + 2];
    if (tid == 0) {
        float* o = newxyz + (size_t)b * S_ * 3;
        o[0] = lx; o[1] = ly; o[2] = lz;
    }
    for (int s = 1; s < S_; ++s) {
        float bestv = -1.0f; int besti = 0;
#pragma unroll
        for (int j = 0; j < 4; ++j) {
            float dx = px[j] - lx, dy = py[j] - ly, dz = pz[j] - lz;
            float d = dx * dx + dy * dy + dz * dz;
            mind[j] = fminf(mind[j], d);
            int p = tid + 1024 * j;
            if (mind[j] > bestv || (mind[j] == bestv && p < besti)) { bestv = mind[j]; besti = p; }
        }
#pragma unroll
        for (int off = 16; off > 0; off >>= 1) {
            float ov = __shfl_down(bestv, (unsigned)off);
            int   oi = __shfl_down(besti, (unsigned)off);
            if (ov > bestv || (ov == bestv && oi < besti)) { bestv = ov; besti = oi; }
        }
        if (lane == 0) { sv[wid] = bestv; si[wid] = besti; }
        __syncthreads();
        if (wid == 0) {
            bestv = sv[lane]; besti = si[lane];
#pragma unroll
            for (int off = 16; off > 0; off >>= 1) {
                float ov = __shfl_down(bestv, (unsigned)off);
                int   oi = __shfl_down(besti, (unsigned)off);
                if (ov > bestv || (ov == bestv && oi < besti)) { bestv = ov; besti = oi; }
            }
            if (lane == 0) {
                const float* pp = pos + ((size_t)b * N_ + besti) * 3;
                slast[0] = pp[0]; slast[1] = pp[1]; slast[2] = pp[2];
                float* o = newxyz + ((size_t)b * S_ + s) * 3;
                o[0] = slast[0]; o[1] = slast[1]; o[2] = slast[2];
            }
        }
        __syncthreads();
        lx = slast[0]; ly = slast[1]; lz = slast[2];
    }
}

// ---------------- ball query: one wave32 per query, first-K in index order ----------
__global__ __launch_bounds__(256) void ballq_kernel(const float* __restrict__ pos,
                                                    const float* __restrict__ newxyz,
                                                    int* __restrict__ idxb,
                                                    int* __restrict__ cntb) {
    int q = blockIdx.x * 8 + (threadIdx.x >> 5);
    int lane = threadIdx.x & 31;
    int b = q >> 10;                       // S_ == 1024
    const float* c = newxyz + (size_t)q * 3;
    float cx = c[0], cy = c[1], cz = c[2];
    const float* pb = pos + (size_t)b * N_ * 3;
    int cnt = 0;
    for (int base = 0; base < N_ && cnt < K_; base += 32) {
        int p = base + lane;
        __builtin_prefetch(pb + (p + 32) * 3, 0, 0);
        float dx = pb[p * 3 + 0] - cx;
        float dy = pb[p * 3 + 1] - cy;
        float dz = pb[p * 3 + 2] - cz;
        bool within = (dx * dx + dy * dy + dz * dz) < R2_;
        unsigned m = (unsigned)__ballot(within);
        int pre = __popc(m & ((1u << lane) - 1u));
        int slot = cnt + pre;
        if (within && slot < K_) idxb[(size_t)q * K_ + slot] = p;
        cnt += __popc(m);
    }
    if (cnt > K_) cnt = K_;
    if (lane == 0) cntb[q] = cnt;
}

// ---------------- fused gather + 3-layer MLP (WMMA bf16) + masked max ---------------
// 1 block (128 thr = 4 waves) processes GPB_ groups; weights staged to LDS once per
// block via async global->LDS (ASYNCcnt). Wave w owns output rows [16w,16w+16).
__global__ __launch_bounds__(128) void groupmlp_kernel(
    const float* __restrict__ x, const float* __restrict__ pos,
    const float* __restrict__ newxyz,
    const int* __restrict__ idxb, const int* __restrict__ cntb,
    const __bf16* __restrict__ w1t, const __bf16* __restrict__ w2t,
    const __bf16* __restrict__ w3t,
    const float* __restrict__ b1, const float* __restrict__ b2,
    const float* __restrict__ b3,
    float* __restrict__ out) {
    __shared__ __align__(16) __bf16 sW1[64][96];
    __shared__ __align__(16) __bf16 sW2[64][64];
    __shared__ __align__(16) __bf16 sW3[128][64];
    __shared__ __align__(16) __bf16 sX[64][96];   // input X0, reused as H2
    __shared__ __align__(16) __bf16 sH[64][64];   // H1
    __shared__ float sP[4][128];

    int tid = threadIdx.x, lane = tid & 31, wid = tid >> 5;
    int nl = lane & 15, h = lane >> 4;

    // ---- stage bf16 weights into LDS once, via async global->LDS b128 copies ----
    {
        unsigned l1 = lds_addr_of(&sW1[0][0]);
        unsigned l2 = lds_addr_of(&sW2[0][0]);
        unsigned l3 = lds_addr_of(&sW3[0][0]);
        const char* g1 = (const char*)w1t;
        const char* g2 = (const char*)w2t;
        const char* g3 = (const char*)w3t;
        for (int i = tid; i < 768; i += 128)   // 12288 B
            async_g2l_b128(l1 + 16u * i, g1 + 16 * i);
        for (int i = tid; i < 512; i += 128)   //  8192 B
            async_g2l_b128(l2 + 16u * i, g2 + 16 * i);
        for (int i = tid; i < 1024; i += 128)  // 16384 B
            async_g2l_b128(l3 + 16u * i, g3 + 16 * i);
        wait_asynccnt0();
    }
    __syncthreads();

    int qbase = blockIdx.x * GPB_;
    for (int g = 0; g < GPB_; ++g) {
        int q = qbase + g;
        int b = q >> 10;
        int cnt = cntb[q];
        __syncthreads();   // previous group's LDS reads complete before rebuild

        // build X0[64][96]: [xyz - center | 64 feats | zero pad]; 2 threads per row
        {
            int r = tid >> 1, hh = tid & 1;
            int p = (r < cnt) ? idxb[(size_t)q * K_ + r] : (N_ - 1);
            const float* c  = newxyz + (size_t)q * 3;
            const float* pp = pos + ((size_t)b * N_ + p) * 3;
            const float* xp = x   + ((size_t)b * N_ + p) * D_;
            if (hh == 0) {
                sX[r][0] = (__bf16)(pp[0] - c[0]);
                sX[r][1] = (__bf16)(pp[1] - c[1]);
                sX[r][2] = (__bf16)(pp[2] - c[2]);
                for (int cc = 3; cc < 48; ++cc) sX[r][cc] = (__bf16)xp[cc - 3];
            } else {
                for (int cc = 48; cc < 67; ++cc) sX[r][cc] = (__bf16)xp[cc - 3];
                for (int cc = 67; cc < 96; ++cc) sX[r][cc] = (__bf16)0.0f;
            }
        }
        __syncthreads();

        const __bf16* arowX = &sX[16 * wid + nl][0];
        const __bf16* arowH = &sH[16 * wid + nl][0];

        // layer 1: [64x96]x[96x64] -> relu -> sH
#pragma unroll
        for (int nt = 0; nt < 4; ++nt) {
            v8f acc = {};
#pragma unroll
            for (int kt = 0; kt < 3; ++kt) {
                v16bf a  = lds_frag_a(arowX, 32 * kt, h);
                v16bf bb = lds_frag_b(&sW1[nt * 16 + nl][0], 32 * kt, h);
                acc = __builtin_amdgcn_wmma_f32_16x16x32_bf16(false, a, false, bb,
                                                              (short)0, acc, false, false);
            }
            float bias = b1[nt * 16 + nl];
#pragma unroll
            for (int r = 0; r < 8; ++r) {
                float v = fmaxf(acc[r] + bias, 0.0f);
                sH[16 * wid + r + 8 * h][nt * 16 + nl] = (__bf16)v;
            }
        }
        __syncthreads();

        // layer 2: [64x64]x[64x64] -> relu -> sX (reused)
#pragma unroll
        for (int nt = 0; nt < 4; ++nt) {
            v8f acc = {};
#pragma unroll
            for (int kt = 0; kt < 2; ++kt) {
                v16bf a  = lds_frag_a(arowH, 32 * kt, h);
                v16bf bb = lds_frag_b(&sW2[nt * 16 + nl][0], 32 * kt, h);
                acc = __builtin_amdgcn_wmma_f32_16x16x32_bf16(false, a, false, bb,
                                                              (short)0, acc, false, false);
            }
            float bias = b2[nt * 16 + nl];
#pragma unroll
            for (int r = 0; r < 8; ++r) {
                float v = fmaxf(acc[r] + bias, 0.0f);
                sX[16 * wid + r + 8 * h][nt * 16 + nl] = (__bf16)v;
            }
        }
        __syncthreads();

        // layer 3: [64x64]x[64x128] + bias, mask invalid rows to 0, max over rows
#pragma unroll
        for (int nt = 0; nt < 8; ++nt) {
            v8f acc = {};
#pragma unroll
            for (int kt = 0; kt < 2; ++kt) {
                v16bf a  = lds_frag_a(arowX, 32 * kt, h);
                v16bf bb = lds_frag_b(&sW3[nt * 16 + nl][0], 32 * kt, h);
                acc = __builtin_amdgcn_wmma_f32_16x16x32_bf16(false, a, false, bb,
                                                              (short)0, acc, false, false);
            }
            float bias = b3[nt * 16 + nl];
            float vmax = -1e30f;
#pragma unroll
            for (int r = 0; r < 8; ++r) {
                int m = 16 * wid + r + 8 * h;
                float v = (m < cnt) ? (acc[r] + bias) : 0.0f;  // mask-then-max, as reference
                vmax = fmaxf(vmax, v);
            }
            vmax = fmaxf(vmax, __shfl_xor(vmax, 16));
            if (lane < 16) sP[wid][nt * 16 + lane] = vmax;
        }
        __syncthreads();
        {
            float v = fmaxf(fmaxf(sP[0][tid], sP[1][tid]), fmaxf(sP[2][tid], sP[3][tid]));
            out[(size_t)q * CO_ + tid] = v;
        }
    }
}

extern "C" void kernel_launch(void* const* d_in, const int* in_sizes, int n_in,
                              void* d_out, int out_size, void* d_ws, size_t ws_size,
                              hipStream_t stream) {
    const float* x   = (const float*)d_in[0];
    const float* pos = (const float*)d_in[1];
    const float* W1  = (const float*)d_in[2];
    const float* b1  = (const float*)d_in[3];
    const float* W2  = (const float*)d_in[4];
    const float* b2  = (const float*)d_in[5];
    const float* W3  = (const float*)d_in[6];
    const float* b3  = (const float*)d_in[7];

    float* out    = (float*)d_out;
    float* newxyz = out + (size_t)B_ * S_ * CO_;   // second tuple output slice

    char* ws = (char*)d_ws;
    __bf16* w1t = (__bf16*)(ws);                    // 12288 B
    __bf16* w2t = (__bf16*)(ws + 12288);            //  8192 B
    __bf16* w3t = (__bf16*)(ws + 20480);            // 16384 B
    int*   cntb = (int*)(ws + 36864);               // 32768 B
    int*   idxb = (int*)(ws + 69632);               // 2 MB

    convw_kernel<<<72, 256, 0, stream>>>(W1, W2, W3, w1t, w2t, w3t);
    fps_kernel<<<B_, 1024, 0, stream>>>(pos, newxyz);
    ballq_kernel<<<(B_ * S_) / 8, 256, 0, stream>>>(pos, newxyz, idxb, cntb);
    groupmlp_kernel<<<(B_ * S_) / GPB_, 128, 0, stream>>>(x, pos, newxyz, idxb, cntb,
                                                          w1t, w2t, w3t, b1, b2, b3, out);
    (void)in_sizes; (void)n_in; (void)out_size; (void)ws_size;
}
